// StructureModule_2989297238042
// MI455X (gfx1250) — matrix operand order repaired and checked by reading, not compiled, not run on hardware
//
#include <hip/hip_runtime.h>
#include <hip/hip_bf16.h>
#include <math.h>

// CDNA5 (gfx1250) wave32 WMMA + TDM implementation of the IPA structure module.
typedef __attribute__((ext_vector_type(16))) __bf16 v16bf;
typedef __attribute__((ext_vector_type(8)))  float  v8f;
typedef __attribute__((ext_vector_type(4)))  unsigned int v4u;
typedef __attribute__((ext_vector_type(4)))  int    v4i;
typedef __attribute__((ext_vector_type(8)))  int    v8i;

#define DI __device__ __forceinline__

namespace {

constexpr int Nn = 768, CSd = 384, CZd = 128, Hh = 12, PQn = 4;
constexpr int PROJ = 1152;           // 192 (q) + 384 (kv) + 144 (qp) + 432 (kvp)
constexpr int CD   = 2112;           // concat width
constexpr float INFV = 100000.f, EPSV = 1e-8f;

// workspace offsets in floats
constexpr size_t OFF_PROJ = 0;
constexpr size_t OFF_QP   = OFF_PROJ + (size_t)Nn * PROJ;     // N x 48 x 3
constexpr size_t OFF_KP   = OFF_QP   + (size_t)Nn * 144;
constexpr size_t OFF_VP   = OFF_KP   + (size_t)Nn * 144;      // N x 96 x 3
constexpr size_t OFF_OPT  = OFF_VP   + (size_t)Nn * 288;      // N x 96 x 3 (global frame o_pt)
constexpr size_t OFF_CAT  = OFF_OPT  + (size_t)Nn * 288;      // N x 2112
constexpr size_t OFF_LOG  = OFF_CAT  + (size_t)Nn * CD;       // H x N x N (logits, then probs)

union AFrag { v16bf v; __bf16 e[16]; };
union CFrag { v8f   v; float  e[8];  };

// ISA 7.12.2: 16-bit A (16x32): lane<16 -> M=lane, K pairs {2r,2r+1} (r<4) / {16+2(r-4),...};
// lanes 16..31 hold the +8 K offset half.
DI int a_kidx(int r, int hi) { return (r < 4 ? 2 * r : 16 + 2 * (r - 4)) + hi * 8; }
// 16-bit B (32x16): lane&15 -> N, same K order as A (pairs {2r,2r+1} (r<4) / {16+2(r-4),...}, lanes>=16 +8).
DI int b_kidx(int r, int hi) { return (r < 4 ? 2 * r : 16 + 2 * (r - 4)) + hi * 8; }

DI float softplusf(float x) { return x > 20.f ? x : log1pf(expf(x)); }

#define WMMA_BF16(accv, Av, Bv) \
  (accv) = __builtin_amdgcn_wmma_f32_16x16x32_bf16(false, (Av), false, (Bv), (short)0, (accv), false, false)

// ---------------------------------------------------------------------------
// Tensor Data Mover: DMA a 32(rows) x 128(cols) fp32 tile of z[i] into LDS.
// D# per ISA ch.8: group0 = count/lds_addr/global_addr/type=2,
// group1 = data_size=4B, tensor_dims 128x768, tile 128x32, stride0=128.
// ---------------------------------------------------------------------------
DI void tdm_load_z_tile(const float* gsrc, unsigned lds_off) {
  const unsigned long long ga = (unsigned long long)(size_t)gsrc;
  v4u g0 = { 1u,                                   // count = 1 valid descriptor
             lds_off,                              // LDS byte address
             (unsigned)ga,                         // global_addr[31:0]
             (unsigned)((ga >> 32) & 0x01FFFFFFu) | 0x80000000u };  // addr[56:32] | type=2
  v8i g1 = { (int)0x00020000,        // workgroup_mask=0, data_size=2 (4 bytes)
             (int)(128u << 16),      // tensor_dim0[15:0]=128 in bits 63:48
             (int)(768u << 16),      // tensor_dim0 hi=0, tensor_dim1[15:0]=768
             (int)(128u << 16),      // tensor_dim1 hi=0, tile_dim0=128
             32,                     // tile_dim1=32, tile_dim2=0
             128,                    // tensor_dim0_stride = 128
             0, 0 };
  v4i g2 = { 0, 0, 0, 0 };
  v4i g3 = { 0, 0, 0, 0 };
#if defined(__clang_major__) && __clang_major__ >= 23
  v8i g4 = { 0, 0, 0, 0, 0, 0, 0, 0 };
  __builtin_amdgcn_tensor_load_to_lds(g0, g1, g2, g3, g4, 0);
#else
  __builtin_amdgcn_tensor_load_to_lds(g0, g1, g2, g3, 0);
#endif
}

// ---------------------------------------------------------------------------
// K1: fused projections  proj = s @ [W_q | W_kv | W_qp | W_kvp] + bias
// grid(48 row-tiles, 72 col-tiles), 1 wave per 16x16 output tile, K=384.
// ---------------------------------------------------------------------------
__global__ void k_proj(const float* __restrict__ s,
                       const float* __restrict__ Wq,  const float* __restrict__ bq,
                       const float* __restrict__ Wkv, const float* __restrict__ bkv,
                       const float* __restrict__ Wqp, const float* __restrict__ bqp,
                       const float* __restrict__ Wkvp,const float* __restrict__ bkvp,
                       float* __restrict__ proj) {
  const int it = blockIdx.x, ct = blockIdx.y;
  const int lane = threadIdx.x, m = lane & 15, hi = lane >> 4, n = m;
  const float* W; const float* bias; int ncols, c0;
  if (ct < 12)      { W = Wq;   bias = bq;   ncols = 192; c0 = ct * 16; }
  else if (ct < 36) { W = Wkv;  bias = bkv;  ncols = 384; c0 = (ct - 12) * 16; }
  else if (ct < 45) { W = Wqp;  bias = bqp;  ncols = 144; c0 = (ct - 36) * 16; }
  else              { W = Wkvp; bias = bkvp; ncols = 432; c0 = (ct - 45) * 16; }
  const int i0 = it * 16;
  CFrag acc; for (int r = 0; r < 8; ++r) acc.e[r] = 0.f;
  for (int kc = 0; kc < CSd; kc += 32) {
    AFrag A, B;
#pragma unroll
    for (int r = 0; r < 8; ++r) {
      const int ka = kc + a_kidx(r, hi);
      const float* srow = s + (size_t)(i0 + m) * CSd + ka;
      A.e[2 * r]     = (__bf16)srow[0];
      A.e[2 * r + 1] = (__bf16)srow[1];
      const int kb = kc + b_kidx(r, hi);
      B.e[2 * r]     = (__bf16)W[(size_t)kb * ncols + c0 + n];
      B.e[2 * r + 1] = (__bf16)W[(size_t)(kb + 1) * ncols + c0 + n];
    }
    WMMA_BF16(acc.v, A.v, B.v);
  }
#pragma unroll
  for (int r = 0; r < 8; ++r) {
    const int mo = r + 8 * hi;
    proj[(size_t)(i0 + mo) * PROJ + ct * 16 + n] = acc.e[r] + bias[c0 + n];
  }
}

// ---------------------------------------------------------------------------
// K2: rigid transform of point projections -> qp / kp / vp (global frame)
// ---------------------------------------------------------------------------
__global__ void k_rigid(const float* __restrict__ proj, const float* __restrict__ rot,
                        const float* __restrict__ trans,
                        float* __restrict__ qp, float* __restrict__ kp, float* __restrict__ vp) {
  const int nres = blockIdx.x;
  const float* R  = rot   + (size_t)nres * 9;
  const float* T  = trans + (size_t)nres * 3;
  const float* pr = proj  + (size_t)nres * PROJ;
  for (int p = threadIdx.x; p < 192; p += blockDim.x) {
    float raw[3];
    if (p < 48)  for (int x = 0; x < 3; ++x) raw[x] = pr[576 + x * 48 + p];
    else { const int pp = p - 48;
           for (int x = 0; x < 3; ++x) raw[x] = pr[720 + x * 144 + pp]; }
    float o[3];
    for (int x = 0; x < 3; ++x)
      o[x] = R[x * 3 + 0] * raw[0] + R[x * 3 + 1] * raw[1] + R[x * 3 + 2] * raw[2] + T[x];
    if (p < 48) {
      for (int x = 0; x < 3; ++x) qp[(size_t)nres * 144 + p * 3 + x] = o[x];
    } else {
      const int pp = p - 48, h = pp / 12, r12 = pp % 12;
      if (r12 < 4) for (int x = 0; x < 3; ++x) kp[(size_t)nres * 144 + (h * 4 + r12) * 3 + x] = o[x];
      else         for (int x = 0; x < 3; ++x) vp[(size_t)nres * 288 + (h * 8 + (r12 - 4)) * 3 + x] = o[x];
    }
  }
}

// ---------------------------------------------------------------------------
// K3: pair bias  logits[h,i,j] = sqrt(1/3) * (z[i,j,:] @ W_b + b_b)[h]
// grid(48,48) (i,j) tiles; 4 waves/block; A rows = 256 flattened cells,
// N = 12 heads padded to 16; first (of two) passes over z.
// ---------------------------------------------------------------------------
__global__ void k_zbias(const float* __restrict__ z, const float* __restrict__ Wb,
                        const float* __restrict__ bb, float* __restrict__ logits) {
  const int it = blockIdx.x, jt = blockIdx.y;
  const int wave = threadIdx.x >> 5, lane = threadIdx.x & 31;
  const int m = lane & 15, hi = lane >> 4, n = m;
  for (int g = wave * 4; g < wave * 4 + 4; ++g) {
    CFrag acc; for (int r = 0; r < 8; ++r) acc.e[r] = 0.f;
    for (int kc = 0; kc < CZd; kc += 32) {
      AFrag A, B;
#pragma unroll
      for (int r = 0; r < 8; ++r) {
        const int cell = g * 16 + m;
        const int i = it * 16 + (cell >> 4), j = jt * 16 + (cell & 15);
        const float* zr = z + ((size_t)i * Nn + j) * CZd;
        const int ka = kc + a_kidx(r, hi);
        A.e[2 * r]     = (__bf16)zr[ka];
        A.e[2 * r + 1] = (__bf16)zr[ka + 1];
        const int kb = kc + b_kidx(r, hi);
        B.e[2 * r]     = (__bf16)(n < Hh ? Wb[(size_t)kb * Hh + n] : 0.f);
        B.e[2 * r + 1] = (__bf16)(n < Hh ? Wb[(size_t)(kb + 1) * Hh + n] : 0.f);
      }
      WMMA_BF16(acc.v, A.v, B.v);
    }
    if (n < Hh) {
      const float sc = 0.57735026919f;  // sqrt(1/3)
#pragma unroll
      for (int r = 0; r < 8; ++r) {
        const int cell = g * 16 + r + 8 * hi;
        const int i = it * 16 + (cell >> 4), j = jt * 16 + (cell & 15);
        logits[((size_t)n * Nn + i) * Nn + j] = sc * (acc.e[r] + bb[n]);
      }
    }
  }
}

// ---------------------------------------------------------------------------
// K4: q.k (K=16 padded to 32) + point term + mask, accumulated into logits.
// grid(H, 48 i-tiles), 1 wave, loop over 48 j-tiles.
// ---------------------------------------------------------------------------
__global__ void k_qk(const float* __restrict__ proj, const float* __restrict__ qp,
                     const float* __restrict__ kp, const float* __restrict__ mask,
                     const float* __restrict__ head_w, float* __restrict__ logits) {
  const int h = blockIdx.x, it = blockIdx.y;
  const int lane = threadIdx.x, m = lane & 15, hi = lane >> 4, n = m;
  const float qk_scale = 0.14433756729740643f;                  // sqrt(1/(3*16))
  const float hw = softplusf(head_w[h]) * 0.13608276348795434f; // sqrt(1/54)
  const int i0 = it * 16;
  AFrag A;
#pragma unroll
  for (int r = 0; r < 8; ++r) {
    const int ka = a_kidx(r, hi);
    const float* qrow = proj + (size_t)(i0 + m) * PROJ + h * 16;
    A.e[2 * r]     = (__bf16)(ka     < 16 ? qrow[ka]     : 0.f);
    A.e[2 * r + 1] = (__bf16)(ka + 1 < 16 ? qrow[ka + 1] : 0.f);
  }
  for (int jt = 0; jt < 48; ++jt) {
    const int j0 = jt * 16;
    AFrag B;
#pragma unroll
    for (int r = 0; r < 8; ++r) {
      const int kb = b_kidx(r, hi);
      const float* krow = proj + (size_t)(j0 + n) * PROJ + 192 + h * 32;
      B.e[2 * r]     = (__bf16)(kb     < 16 ? krow[kb]     : 0.f);
      B.e[2 * r + 1] = (__bf16)(kb + 1 < 16 ? krow[kb + 1] : 0.f);
    }
    CFrag acc; for (int r = 0; r < 8; ++r) acc.e[r] = 0.f;
    WMMA_BF16(acc.v, A.v, B.v);
    const int j = j0 + n;
    const float mj = mask[j];
#pragma unroll
    for (int r = 0; r < 8; ++r) {
      const int i = i0 + r + 8 * hi;
      float d2 = 0.f;
#pragma unroll
      for (int p = 0; p < PQn; ++p)
#pragma unroll
        for (int x = 0; x < 3; ++x) {
          const float d = qp[(size_t)i * 144 + (h * 4 + p) * 3 + x]
                        - kp[(size_t)j * 144 + (h * 4 + p) * 3 + x];
          d2 += d * d;
        }
      const float val = qk_scale * acc.e[r] - 0.5f * hw * d2 + INFV * (mask[i] * mj - 1.f);
      logits[((size_t)h * Nn + i) * Nn + j] += val;
    }
  }
}

// ---------------------------------------------------------------------------
// K5: weighted softmax over j.  weights = exp(ss) - 0.99
// ---------------------------------------------------------------------------
__global__ void k_softmax(const float* __restrict__ ss, float* __restrict__ logits) {
  const int row = blockIdx.x;          // h*N + i
  const int i = row % Nn;
  float* L = logits + (size_t)row * Nn;
  __shared__ float red[256];
  const int t = threadIdx.x;
  float mx = -INFINITY;
  for (int j = t; j < Nn; j += 256) mx = fmaxf(mx, L[j]);
  red[t] = mx; __syncthreads();
  for (int s = 128; s > 0; s >>= 1) { if (t < s) red[t] = fmaxf(red[t], red[t + s]); __syncthreads(); }
  mx = red[0]; __syncthreads();
  float sum = 0.f;
  for (int j = t; j < Nn; j += 256) {
    const float w = expf(ss[(size_t)i * Nn + j]) - 0.99f;
    const float e = expf(L[j] - mx) * w;
    L[j] = e; sum += e;
  }
  red[t] = sum; __syncthreads();
  for (int s = 128; s > 0; s >>= 1) { if (t < s) red[t] += red[t + s]; __syncthreads(); }
  const float inv = 1.f / red[0];
  for (int j = t; j < Nn; j += 256) L[j] *= inv;
}

// ---------------------------------------------------------------------------
// K6: o = a @ v (-> cat), o_pt_global = a @ vp (-> scratch).
// grid(48 i-tiles, H, 3 col-tiles): ct0 = v (16), ct1/2 = vp 24 cols padded.
// ---------------------------------------------------------------------------
__global__ void k_av(const float* __restrict__ probs, const float* __restrict__ proj,
                     const float* __restrict__ vp, float* __restrict__ cat,
                     float* __restrict__ optg) {
  const int it = blockIdx.x, h = blockIdx.y, ct = blockIdx.z;
  const int lane = threadIdx.x, m = lane & 15, hi = lane >> 4, n = m;
  const int i0 = it * 16;
  CFrag acc; for (int r = 0; r < 8; ++r) acc.e[r] = 0.f;
  for (int kc = 0; kc < Nn; kc += 32) {
    AFrag A, B;
#pragma unroll
    for (int r = 0; r < 8; ++r) {
      const int ka = kc + a_kidx(r, hi);
      const float* ar = probs + ((size_t)h * Nn + i0 + m) * Nn;
      A.e[2 * r]     = (__bf16)ar[ka];
      A.e[2 * r + 1] = (__bf16)ar[ka + 1];
      const int kb = kc + b_kidx(r, hi);
#pragma unroll
      for (int u = 0; u < 2; ++u) {
        const int j = kb + u;
        float bv;
        if (ct == 0) bv = proj[(size_t)j * PROJ + 192 + h * 32 + 16 + n];
        else { const int col = (ct == 1 ? n : 16 + n);
               bv = (col < 24) ? vp[(size_t)j * 288 + h * 24 + col] : 0.f; }
        B.e[2 * r + u] = (__bf16)bv;
      }
    }
    WMMA_BF16(acc.v, A.v, B.v);
  }
#pragma unroll
  for (int r = 0; r < 8; ++r) {
    const int i = i0 + r + 8 * hi;
    if (ct == 0) cat[(size_t)i * CD + h * 16 + n] = acc.e[r];
    else { const int col = (ct == 1 ? n : 16 + n);
           if (col < 24) optg[(size_t)i * 288 + h * 24 + col] = acc.e[r]; }
  }
}

// ---------------------------------------------------------------------------
// K7: inverse rigid + norms for o_pt -> cat columns [192, 576)
// ---------------------------------------------------------------------------
__global__ void k_optfin(const float* __restrict__ optg, const float* __restrict__ rot,
                         const float* __restrict__ trans, float* __restrict__ cat) {
  const int nres = blockIdx.x;
  const int t = threadIdx.x;  // h*8+p in [0,96)
  if (t >= 96) return;
  const float* R = rot + (size_t)nres * 9;
  const float* T = trans + (size_t)nres * 3;
  float g[3];
  for (int x = 0; x < 3; ++x) g[x] = optg[(size_t)nres * 288 + t * 3 + x] - T[x];
  float l[3];
  for (int x = 0; x < 3; ++x) l[x] = R[0 * 3 + x] * g[0] + R[1 * 3 + x] * g[1] + R[2 * 3 + x] * g[2];
  const float nrm = sqrtf(l[0] * l[0] + l[1] * l[1] + l[2] * l[2] + EPSV);
  float* crow = cat + (size_t)nres * CD;
  for (int x = 0; x < 3; ++x) crow[192 + x * 96 + t] = l[x];
  crow[480 + t] = nrm;
}

// ---------------------------------------------------------------------------
// K8: o_pair[i] = a[:,i,:] @ z[i]   (M=12 heads pad 16, K=768, N=128)
// grid(768 residues), block = 8 waves (one per 16-wide cz tile).
// Second (last) pass over z, staged through LDS by the Tensor Data Mover
// with double buffering: wave 0 issues tensor_load_to_lds for chunk kc+1
// while all 8 waves run WMMA on chunk kc; s_wait_tensorcnt + barrier rotate.
// ---------------------------------------------------------------------------
__global__ void k_opair(const float* __restrict__ probs, const float* __restrict__ z,
                        float* __restrict__ cat) {
  const int i = blockIdx.x;
  const int wave = threadIdx.x >> 5, lane = threadIdx.x & 31;
  const int m = lane & 15, hi = lane >> 4, n = m;
  const int cz0 = wave * 16;
  __shared__ float zbuf[2][32 * CZd];   // 2 x 16 KB double buffer
  const float* zi = z + (size_t)i * Nn * CZd;

  if (wave == 0) {
    tdm_load_z_tile(zi, (unsigned)(size_t)&zbuf[0][0]);
    __builtin_amdgcn_s_wait_tensorcnt(0);
  }
  __syncthreads();

  CFrag acc; for (int r = 0; r < 8; ++r) acc.e[r] = 0.f;
  const float* arow = probs + ((size_t)m * Nn + i) * Nn;
  for (int kc = 0; kc < Nn; kc += 32) {
    const int buf = (kc >> 5) & 1;
    if (wave == 0 && kc + 32 < Nn)
      tdm_load_z_tile(zi + (size_t)(kc + 32) * CZd, (unsigned)(size_t)&zbuf[buf ^ 1][0]);
    AFrag A, B;
#pragma unroll
    for (int r = 0; r < 8; ++r) {
      const int ka = kc + a_kidx(r, hi);
      A.e[2 * r]     = (__bf16)(m < Hh ? arow[ka]     : 0.f);
      A.e[2 * r + 1] = (__bf16)(m < Hh ? arow[ka + 1] : 0.f);
      const int kb = b_kidx(r, hi);
      B.e[2 * r]     = (__bf16)zbuf[buf][(size_t)kb * CZd + cz0 + n];
      B.e[2 * r + 1] = (__bf16)zbuf[buf][(size_t)(kb + 1) * CZd + cz0 + n];
    }
    WMMA_BF16(acc.v, A.v, B.v);
    if (wave == 0) __builtin_amdgcn_s_wait_tensorcnt(0);
    __syncthreads();
  }
#pragma unroll
  for (int r = 0; r < 8; ++r) {
    const int h = r + 8 * hi;
    if (h < Hh) cat[(size_t)i * CD + 576 + h * CZd + cz0 + n] = acc.e[r];
  }
}

// ---------------------------------------------------------------------------
// K9: out = cat @ W_out + b_out   (768 x 2112 x 384)
// ---------------------------------------------------------------------------
__global__ void k_out(const float* __restrict__ cat, const float* __restrict__ Wout,
                      const float* __restrict__ bout, float* __restrict__ out) {
  const int it = blockIdx.x, ctile = blockIdx.y;
  const int lane = threadIdx.x, m = lane & 15, hi = lane >> 4, n = m;
  const int i0 = it * 16, c0 = ctile * 16;
  CFrag acc; for (int r = 0; r < 8; ++r) acc.e[r] = 0.f;
  for (int kc = 0; kc < CD; kc += 32) {
    if (kc + 32 < CD)
      __builtin_prefetch(&Wout[(size_t)(kc + 32) * CSd + c0 + n], 0, 1);  // global_prefetch_b8
    AFrag A, B;
#pragma unroll
    for (int r = 0; r < 8; ++r) {
      const int ka = kc + a_kidx(r, hi);
      const float* crow = cat + (size_t)(i0 + m) * CD;
      A.e[2 * r]     = (__bf16)crow[ka];
      A.e[2 * r + 1] = (__bf16)crow[ka + 1];
      const int kb = kc + b_kidx(r, hi);
      B.e[2 * r]     = (__bf16)Wout[(size_t)kb * CSd + c0 + n];
      B.e[2 * r + 1] = (__bf16)Wout[(size_t)(kb + 1) * CSd + c0 + n];
    }
    WMMA_BF16(acc.v, A.v, B.v);
  }
#pragma unroll
  for (int r = 0; r < 8; ++r) {
    const int i = i0 + r + 8 * hi;
    out[(size_t)i * CSd + c0 + n] = acc.e[r] + bout[c0 + n];
  }
}

}  // namespace

extern "C" void kernel_launch(void* const* d_in, const int* in_sizes, int n_in,
                              void* d_out, int out_size, void* d_ws, size_t ws_size,
                              hipStream_t stream) {
  const float* s      = (const float*)d_in[0];
  const float* z      = (const float*)d_in[1];
  const float* rot    = (const float*)d_in[2];
  const float* trans  = (const float*)d_in[3];
  const float* mask   = (const float*)d_in[4];
  const float* ss     = (const float*)d_in[5];
  const float* W_q    = (const float*)d_in[6];
  const float* b_q    = (const float*)d_in[7];
  const float* W_kv   = (const float*)d_in[8];
  const float* b_kv   = (const float*)d_in[9];
  const float* W_qp   = (const float*)d_in[10];
  const float* b_qp   = (const float*)d_in[11];
  const float* W_kvp  = (const float*)d_in[12];
  const float* b_kvp  = (const float*)d_in[13];
  const float* W_b    = (const float*)d_in[14];
  const float* b_b    = (const float*)d_in[15];
  const float* head_w = (const float*)d_in[16];
  const float* W_out  = (const float*)d_in[17];
  const float* b_out  = (const float*)d_in[18];
  float* out = (float*)d_out;

  float* ws    = (float*)d_ws;
  float* proj  = ws + OFF_PROJ;
  float* qp    = ws + OFF_QP;
  float* kp    = ws + OFF_KP;
  float* vp    = ws + OFF_VP;
  float* optg  = ws + OFF_OPT;
  float* cat   = ws + OFF_CAT;
  float* logit = ws + OFF_LOG;

  k_proj<<<dim3(48, 72), 32, 0, stream>>>(s, W_q, b_q, W_kv, b_kv, W_qp, b_qp, W_kvp, b_kvp, proj);
  k_rigid<<<768, 64, 0, stream>>>(proj, rot, trans, qp, kp, vp);
  k_zbias<<<dim3(48, 48), 128, 0, stream>>>(z, W_b, b_b, logit);
  k_qk<<<dim3(12, 48), 32, 0, stream>>>(proj, qp, kp, mask, head_w, logit);
  k_softmax<<<12 * 768, 256, 0, stream>>>(ss, logit);
  k_av<<<dim3(48, 12, 3), 32, 0, stream>>>(logit, proj, vp, cat, optg);
  k_optfin<<<768, 96, 0, stream>>>(optg, rot, trans, cat);
  k_opair<<<768, 256, 0, stream>>>(logit, z, cat);
  k_out<<<dim3(48, 24), 32, 0, stream>>>(cat, W_out, b_out, out);
}
